// MesBceWasLoss_33191507263733
// MI455X (gfx1250) — matrix operand order repaired and checked by reading, not compiled, not run on hardware
//
#include <hip/hip_runtime.h>

#define SIZE  256
#define NKP   68
#define KPAD  96      // 68 padded to 3 K-tiles of 32 (zeros)
#define BATCH 16
#define SK_ITERS 50

typedef __attribute__((ext_vector_type(16))) __bf16         v16bf;
typedef __attribute__((ext_vector_type(8)))  float          v8f;
typedef __attribute__((ext_vector_type(8)))  unsigned short ushort8;
typedef __attribute__((ext_vector_type(16))) unsigned short ushort16;

__device__ __forceinline__ unsigned short f32_to_bf16(float f) {
    unsigned int u = __float_as_uint(f);
    unsigned int r = (u + 0x7FFFu + ((u >> 16) & 1u)) >> 16;   // RNE
    return (unsigned short)r;
}
__device__ __forceinline__ float bf16_to_f32(unsigned short h) {
    return __uint_as_float(((unsigned int)h) << 16);
}

// ---------------------------------------------------------------------------
// Kernel 1: build bf16 Gaussian factor matrices, K-contiguous layout [b][s][KPAD]
// gx[b][s][n] = exp(-(s - px_x*255)^2 / (2*sigma^2)),  gy likewise with y coord
// ---------------------------------------------------------------------------
__global__ void build_gauss(const float* __restrict__ pred,
                            const float* __restrict__ targ,
                            unsigned short* __restrict__ gyp,
                            unsigned short* __restrict__ gxp,
                            unsigned short* __restrict__ gyt,
                            unsigned short* __restrict__ gxt) {
    __shared__ float pc[NKP * 2];
    __shared__ float tc[NKP * 2];
    const int b = blockIdx.x;
    const int s = threadIdx.x;                 // 0..255
    for (int i = threadIdx.x; i < NKP * 2; i += blockDim.x) {
        pc[i] = pred[b * NKP * 2 + i];
        tc[i] = targ[b * NKP * 2 + i];
    }
    __syncthreads();

    const float inv = 1.0f / (2.0f * 4.0f * 4.0f);   // 1/(2*sigma^2), sigma=4
    const float xs = (float)s;
    const size_t row = ((size_t)b * SIZE + s) * KPAD;
    for (int k = 0; k < KPAD; ++k) {
        float vxp = 0.f, vyp = 0.f, vxt = 0.f, vyt = 0.f;
        if (k < NKP) {
            float dxp = xs - pc[2 * k + 0] * 255.0f;
            float dyp = xs - pc[2 * k + 1] * 255.0f;
            float dxt = xs - tc[2 * k + 0] * 255.0f;
            float dyt = xs - tc[2 * k + 1] * 255.0f;
            vxp = __expf(-dxp * dxp * inv);
            vyp = __expf(-dyp * dyp * inv);
            vxt = __expf(-dxt * dxt * inv);
            vyt = __expf(-dyt * dyt * inv);
        }
        gxp[row + k] = f32_to_bf16(vxp);
        gyp[row + k] = f32_to_bf16(vyp);
        gxt[row + k] = f32_to_bf16(vxt);
        gyt[row + k] = f32_to_bf16(vyt);
    }
}

// ---------------------------------------------------------------------------
// Kernel 2: per-batch heatmap sums via separability:
//   S[b] = sum_n (sum_s gx[b][s][n]) * (sum_s gy[b][s][n])
// blockIdx.x = which*16 + b   (which: 0=pred, 1=target)
// ---------------------------------------------------------------------------
__global__ void heat_sums(const unsigned short* __restrict__ gyp,
                          const unsigned short* __restrict__ gxp,
                          const unsigned short* __restrict__ gyt,
                          const unsigned short* __restrict__ gxt,
                          float* __restrict__ S) {
    const int b     = blockIdx.x & 15;
    const int which = blockIdx.x >> 4;
    const unsigned short* gy = which ? gyt : gyp;
    const unsigned short* gx = which ? gxt : gxp;
    const int t = threadIdx.x;                 // 128 threads
    float partial = 0.f;
    if (t < KPAD) {
        float sx = 0.f, sy = 0.f;
        size_t base = (size_t)b * SIZE * KPAD + t;
        for (int s = 0; s < SIZE; ++s) {
            sx += bf16_to_f32(gx[base + (size_t)s * KPAD]);
            sy += bf16_to_f32(gy[base + (size_t)s * KPAD]);
        }
        partial = sx * sy;                     // zero automatically for padded n
    }
    __shared__ float red[128];
    red[t] = partial;
    __syncthreads();
    for (int off = 64; off > 0; off >>= 1) {
        if (t < off) red[t] += red[t + off];
        __syncthreads();
    }
    if (t == 0) S[which * BATCH + b] = red[0];
}

// ---------------------------------------------------------------------------
// Kernel 3: fused WMMA heatmap GEMM + BCE reduction.
// One wave = one 16x16 output tile; computes pred AND target tiles (K=96 -> 3+3
// v_wmma_f32_16x16x32_bf16), then BCE contribution, wave-reduce, global atomic.
// ---------------------------------------------------------------------------
__device__ __forceinline__ v16bf load_A(const unsigned short* __restrict__ row,
                                        int kt, int kh) {
    // A 16x32 layout: lane half kh -> contiguous K = kt*32 + kh*16 .. +15 (matches B)
    const unsigned short* p = row + kt * 32 + kh * 16;
    ushort8 lo = *reinterpret_cast<const ushort8*>(p);
    ushort8 hi = *reinterpret_cast<const ushort8*>(p + 8);
    ushort16 u;
#pragma unroll
    for (int i = 0; i < 8; ++i) { u[i] = lo[i]; u[8 + i] = hi[i]; }
    return __builtin_bit_cast(v16bf, u);
}
__device__ __forceinline__ v16bf load_B(const unsigned short* __restrict__ row,
                                        int kt, int kh) {
    // B 32x16 column-major layout, this code's k order (same as A): lane half kh -> contiguous K = kt*32 + kh*16 .. +15
    const unsigned short* p = row + kt * 32 + kh * 16;
    ushort16 u = *reinterpret_cast<const ushort16*>(p);
    return __builtin_bit_cast(v16bf, u);
}

__global__ void wmma_bce(const unsigned short* __restrict__ gyp,
                         const unsigned short* __restrict__ gxp,
                         const unsigned short* __restrict__ gyt,
                         const unsigned short* __restrict__ gxt,
                         const float* __restrict__ S,
                         float* __restrict__ acc) {
    const int lane = threadIdx.x & 31;
    const int wid  = threadIdx.x >> 5;
    const int tile = blockIdx.x * 8 + wid;     // 4096 tiles total
    const int b  = tile >> 8;
    const int ty = (tile >> 4) & 15;
    const int tx = tile & 15;

    const int mrow = lane & 15;                // A row index / B column index
    const int kh   = lane >> 4;                // K-half selector

    const size_t arow = ((size_t)b * SIZE + ty * 16 + mrow) * KPAD;
    const size_t brow = ((size_t)b * SIZE + tx * 16 + mrow) * KPAD;

    v8f cp = {}; v8f ct = {};
#pragma unroll
    for (int kt = 0; kt < 3; ++kt) {
        v16bf ap = load_A(gyp + arow, kt, kh);
        v16bf bp = load_B(gxp + brow, kt, kh);
        cp = __builtin_amdgcn_wmma_f32_16x16x32_bf16(false, ap, false, bp,
                                                     (short)0, cp, false, false);
        v16bf at = load_A(gyt + arow, kt, kh);
        v16bf bt = load_B(gxt + brow, kt, kh);
        ct = __builtin_amdgcn_wmma_f32_16x16x32_bf16(false, at, false, bt,
                                                     (short)0, ct, false, false);
    }

    const float inv_sp = 1.0f / (S[b] + 1e-8f);
    const float inv_st = 1.0f / S[BATCH + b];
    float contrib = 0.f;
#pragma unroll
    for (int r = 0; r < 8; ++r) {
        float p = cp[r] * inv_sp;
        float t = ct[r] * inv_st;
        float lp   = fmaxf(__logf(p),    -100.0f);   // log clamp at -100
        float l1mp = fmaxf(log1pf(-p),   -100.0f);
        contrib += t * lp + (1.0f - t) * l1mp;
    }
#pragma unroll
    for (int off = 16; off > 0; off >>= 1)
        contrib += __shfl_xor(contrib, off, 32);
    if (lane == 0) atomicAdd(&acc[0], contrib);
}

// ---------------------------------------------------------------------------
// Kernel 4: Sinkhorn entropic OT, one block per batch, all state in LDS.
// ---------------------------------------------------------------------------
__global__ void sinkhorn(const float* __restrict__ pred,
                         const float* __restrict__ targ,
                         float* __restrict__ acc) {
    const float eps     = 0.01f;
    const float inv_eps = 100.0f;
    const float log_a   = -4.21950770518f;     // -ln(68)

    __shared__ float xc[NKP * 2];
    __shared__ float yc[NKP * 2];
    __shared__ float C[NKP][NKP];
    __shared__ float f[NKP], g[NKP];
    __shared__ float red[128];

    const int b = blockIdx.x;
    const int t = threadIdx.x;                 // 128 threads
    for (int i = t; i < NKP * 2; i += 128) {
        xc[i] = pred[b * NKP * 2 + i];
        yc[i] = targ[b * NKP * 2 + i];
    }
    if (t < NKP) { f[t] = 0.f; g[t] = 0.f; }
    __syncthreads();

    for (int idx = t; idx < NKP * NKP; idx += 128) {
        int i = idx / NKP, j = idx % NKP;
        float dx = xc[2 * i]     - yc[2 * j];
        float dy = xc[2 * i + 1] - yc[2 * j + 1];
        C[i][j] = dx * dx + dy * dy;
    }
    __syncthreads();

    for (int it = 0; it < SK_ITERS; ++it) {
        if (t < NKP) {
            float m = -1e30f;
            for (int j = 0; j < NKP; ++j)
                m = fmaxf(m, (g[j] - C[t][j]) * inv_eps + log_a);
            float ssum = 0.f;
            for (int j = 0; j < NKP; ++j)
                ssum += __expf((g[j] - C[t][j]) * inv_eps + log_a - m);
            f[t] = -eps * (m + __logf(ssum));
        }
        __syncthreads();
        if (t < NKP) {
            float m = -1e30f;
            for (int i = 0; i < NKP; ++i)
                m = fmaxf(m, (f[i] - C[i][t]) * inv_eps + log_a);
            float ssum = 0.f;
            for (int i = 0; i < NKP; ++i)
                ssum += __expf((f[i] - C[i][t]) * inv_eps + log_a - m);
            g[t] = -eps * (m + __logf(ssum));
        }
        __syncthreads();
    }

    float cost = 0.f;
    for (int idx = t; idx < NKP * NKP; idx += 128) {
        int i = idx / NKP, j = idx % NKP;
        float lP = (f[i] + g[j] - C[i][j]) * inv_eps + 2.0f * log_a;
        cost += __expf(lP) * C[i][j];
    }
    red[t] = cost;
    __syncthreads();
    for (int off = 64; off > 0; off >>= 1) {
        if (t < off) red[t] += red[t + off];
        __syncthreads();
    }
    if (t == 0) atomicAdd(&acc[1], red[0]);
}

// ---------------------------------------------------------------------------
__global__ void init_acc(float* __restrict__ acc) {
    if (threadIdx.x < 2) acc[threadIdx.x] = 0.f;
}

__global__ void finalize(const float* __restrict__ acc, float* __restrict__ out) {
    if (threadIdx.x == 0) {
        float bce_mean = -acc[0] / (float)(BATCH * SIZE * SIZE);
        float was_mean =  acc[1] / (float)BATCH;
        out[0] = bce_mean * 1000000.0f + was_mean * 2000.0f;
    }
}

// ---------------------------------------------------------------------------
extern "C" void kernel_launch(void* const* d_in, const int* in_sizes, int n_in,
                              void* d_out, int out_size, void* d_ws, size_t ws_size,
                              hipStream_t stream) {
    const float* pred = (const float*)d_in[0];
    const float* targ = (const float*)d_in[1];

    const size_t MAT = (size_t)BATCH * SIZE * KPAD;    // bf16 elems per matrix
    unsigned short* gyp = (unsigned short*)d_ws;
    unsigned short* gxp = gyp + MAT;
    unsigned short* gyt = gxp + MAT;
    unsigned short* gxt = gyt + MAT;
    float* S   = (float*)(gxt + MAT);                  // 32 floats: Sp[16], St[16]
    float* acc = S + 32;                               // acc[0]=bce sum, acc[1]=was sum
    float* out = (float*)d_out;

    init_acc  <<<1,   32,  0, stream>>>(acc);
    build_gauss<<<BATCH, 256, 0, stream>>>(pred, targ, gyp, gxp, gyt, gxt);
    heat_sums <<<32,  128, 0, stream>>>(gyp, gxp, gyt, gxt, S);
    wmma_bce  <<<512, 256, 0, stream>>>(gyp, gxp, gyt, gxt, S, acc);  // 4096 tiles, 8 waves/blk
    sinkhorn  <<<BATCH, 128, 0, stream>>>(pred, targ, acc);
    finalize  <<<1,   1,   0, stream>>>(acc, out);
}
